// GIN4_9294309228817
// MI455X (gfx1250) — matrix operand order repaired and checked by reading, not compiled, not run on hardware
//
#include <hip/hip_runtime.h>

// ---------------------------------------------------------------------------
// GINE 3-layer GNN for MI455X (gfx1250, wave32, WMMA, bf16 activations)
//   layer:  agg = x + scatter_add(relu(x[src] + ea*We + be), dst)
//           h   = relu(agg @ W + b)
//   Layers 2/3: bf16 h/agg, global_atomic_pk_add_bf16 scatter,
//               LDS-staged A tiles, v_wmma_f32_16x16x32_bf16 GEMM,
//               LDS-transposed epilogue, dual-write (h + next agg) b128 stores.
// ---------------------------------------------------------------------------

typedef __attribute__((ext_vector_type(16))) __bf16          v16bf;
typedef __attribute__((ext_vector_type(8)))  unsigned int    v8u;
typedef __attribute__((ext_vector_type(8)))  float           v8f;

__device__ __forceinline__ unsigned short f2bf(float f) {
    unsigned u = __float_as_uint(f);
    u += 0x7FFFu + ((u >> 16) & 1u);   // round-to-nearest-even
    return (unsigned short)(u >> 16);
}
__device__ __forceinline__ float bf2f(unsigned short h) {
    return __uint_as_float(((unsigned)h) << 16);
}
// CDNA5 packed bf16 atomic add (no-return -> STOREcnt); ISA 15.18.3 op 90
__device__ __forceinline__ void atomic_pk_add_bf16(unsigned short* addr, unsigned packed) {
    asm volatile("global_atomic_pk_add_bf16 %0, %1, off"
                 :: "v"((unsigned long long)(uintptr_t)addr), "v"(packed)
                 : "memory");
}

// ---------------- generic small kernels ----------------

__global__ void copy_f32(float* __restrict__ dst, const float* __restrict__ src, int n) {
    int i = blockIdx.x * blockDim.x + threadIdx.x;
    if (i < n) dst[i] = src[i];
}
__global__ void zero_f32(float* __restrict__ p, int n) {
    int i = blockIdx.x * blockDim.x + threadIdx.x;
    if (i < n) p[i] = 0.0f;
}

// ---------------- layer 1: in = 7 (f32 path, tiny) ----------------

__global__ void edge_scatter7(const float* __restrict__ x,
                              const int* __restrict__ src,
                              const int* __restrict__ dst,
                              const float* __restrict__ ea,
                              const float* __restrict__ We,
                              const float* __restrict__ be,
                              float* __restrict__ agg, int E) {
    int e = blockIdx.x * blockDim.x + threadIdx.x;
    if (e >= E) return;
    int s = src[e], d = dst[e];
    float a = ea[e];
    const float* xs = x + (long)s * 7;
    float* ad = agg + (long)d * 7;
#pragma unroll
    for (int f = 0; f < 7; ++f) {
        float m = fmaxf(xs[f] + a * We[f] + be[f], 0.0f);
        unsafeAtomicAdd(&ad[f], m);
    }
}

// h(bf16) = relu(agg7 @ W + b); dual-writes next layer's agg buffer
__global__ void gemm7_relu(const float* __restrict__ agg,
                           const float* __restrict__ W,   // [7,128]
                           const float* __restrict__ b,   // [128]
                           unsigned short* __restrict__ h,
                           unsigned short* __restrict__ agg_out, int N) {
    int t = blockIdx.x * blockDim.x + threadIdx.x;
    if (t >= N * 128) return;
    int o = t & 127;
    int n = t >> 7;
    const float* ar = agg + (long)n * 7;
    float acc = b[o];
#pragma unroll
    for (int f = 0; f < 7; ++f) acc = fmaf(ar[f], W[f * 128 + o], acc);
    unsigned short v = f2bf(fmaxf(acc, 0.0f));
    h[t] = v;
    agg_out[t] = v;
}

// ---------------- layers 2/3: in = 128 (bf16 path) ----------------

// one wave32 per edge; lane handles 4 features: 8B gather + 2 pk-bf16 atomics
__global__ void edge_scatter128_k(const unsigned short* __restrict__ x,
                                  const int* __restrict__ src,
                                  const int* __restrict__ dst,
                                  const float* __restrict__ ea,
                                  const float* __restrict__ We,
                                  const float* __restrict__ be,
                                  unsigned short* __restrict__ agg, int E) {
    int lane = threadIdx.x & 31;
    int wv   = threadIdx.x >> 5;
    int e    = blockIdx.x * 8 + wv;
    if (e >= E) return;
    int s = src[e], d = dst[e];
    float a = ea[e];
    int f = lane * 4;
    uint2  xv  = *(const uint2*) (x  + (long)s * 128 + f);
    float4 wv4 = *(const float4*)(We + f);
    float4 bv4 = *(const float4*)(be + f);
    float x0 = bf2f((unsigned short)(xv.x & 0xffff));
    float x1 = bf2f((unsigned short)(xv.x >> 16));
    float x2 = bf2f((unsigned short)(xv.y & 0xffff));
    float x3 = bf2f((unsigned short)(xv.y >> 16));
    float m0 = fmaxf(fmaf(a, wv4.x, x0) + bv4.x, 0.0f);
    float m1 = fmaxf(fmaf(a, wv4.y, x1) + bv4.y, 0.0f);
    float m2 = fmaxf(fmaf(a, wv4.z, x2) + bv4.z, 0.0f);
    float m3 = fmaxf(fmaf(a, wv4.w, x3) + bv4.w, 0.0f);
    unsigned p0 = ((unsigned)f2bf(m1) << 16) | f2bf(m0);
    unsigned p1 = ((unsigned)f2bf(m3) << 16) | f2bf(m2);
    unsigned short* ad = agg + (long)d * 128 + f;
    atomic_pk_add_bf16(ad + 0, p0);
    atomic_pk_add_bf16(ad + 2, p1);
}

// Pre-swizzle W [128,128] f32 into bf16 B-fragment layout:
//   index = ((cb*4 + kt)*32 + lane)*16 + e
//   lane<16:  N = cb*16 + lane,     K = kt*32 + (e&7) + 16*(e>>3)
//   lane>=16: N = cb*16 + lane-16,  K = kt*32 + 8 + (e&7) + 16*(e>>3)
__global__ void stage_w_bf16(const float* __restrict__ W, unsigned short* __restrict__ Wb) {
    int t = blockIdx.x * blockDim.x + threadIdx.x;
    if (t >= 128 * 128) return;
    int e    = t & 15;
    int lane = (t >> 4) & 31;
    int kt   = (t >> 9) & 3;
    int cb   = t >> 11;
    int n = cb * 16 + (lane & 15);
    int k = kt * 32 + ((lane >> 4) * 8) + (e & 7) + ((e >> 3) * 16);
    Wb[t] = f2bf(W[k * 128 + n]);
}

// h = relu(agg @ W + b), all bf16 in/out, f32 accum via WMMA.
// Block = 256 thr = 8 waves; block covers 16 rows, wave w covers cols [w*16,+16).
// A tile (16x128 bf16 = 4KB) staged in LDS; output tile routed back through the
// same LDS buffer; coalesced b128 stores to hout and (optionally) agg_out.
// agg_out may alias agg: each block only overwrites rows it already consumed.
__global__ void gemm128_wmma_relu(const unsigned short* __restrict__ agg,
                                  const unsigned short* __restrict__ Wb,
                                  const float* __restrict__ bias,
                                  unsigned short* __restrict__ hout,
                                  unsigned short* __restrict__ agg_out, int N) {
    __shared__ uint4 ldsT4[16 * 128 / 8];            // 4KB: A tile, then out tile
    unsigned short* ldsT = (unsigned short*)ldsT4;

    int t  = threadIdx.x;
    int m0 = blockIdx.x * 16;
    int row16 = t >> 4;                               // 0..15
    int col8  = (t & 15) * 8;                         // 0,8,..,120
    {   // cooperative A-tile load: 256 thr x 16B = 4KB
        int grow = m0 + row16;
        if (grow >= N) grow = N - 1;
        *(uint4*)(ldsT + row16 * 128 + col8) = *(const uint4*)(agg + (long)grow * 128 + col8);
    }
    __syncthreads();

    int lane = t & 31;
    int wv   = t >> 5;
    int kb   = (lane >> 4) * 8;                       // A K-offset per half-wave
    const unsigned short* arow = ldsT + (lane & 15) * 128 + kb;

    v8f c = {};
#pragma unroll
    for (int kt = 0; kt < 4; ++kt) {
        uint4 lo = *(const uint4*)(arow + kt * 32);        // K = kb+0..7
        uint4 hi = *(const uint4*)(arow + kt * 32 + 16);   // K = kb+16..23
        v8u au;
        au[0] = lo.x; au[1] = lo.y; au[2] = lo.z; au[3] = lo.w;
        au[4] = hi.x; au[5] = hi.y; au[6] = hi.z; au[7] = hi.w;
        v16bf a  = __builtin_bit_cast(v16bf, au);
        v16bf bm = *(const v16bf*)(Wb + ((wv * 4 + kt) * 32 + lane) * 16);
        c = __builtin_amdgcn_wmma_f32_16x16x32_bf16(
                false, a, false, bm, (short)0, c, false, false);
    }
    __syncthreads();   // A reads done; reuse ldsT for the output tile

    // C/D layout: vgpr r, lanes 0-15 -> tile-row r, lanes 16-31 -> tile-row 8+r
    int n     = wv * 16 + (lane & 15);                // output column 0..127
    int rbase = (lane >> 4) * 8;                      // tile-row base
    float bv  = bias[n];
#pragma unroll
    for (int r = 0; r < 8; ++r)
        ldsT[(rbase + r) * 128 + n] = f2bf(fmaxf(c[r] + bv, 0.0f));
    __syncthreads();

    {   // coalesced tile store: 256 thr x 16B (x2 if feeding next layer's agg)
        int m = m0 + row16;
        if (m < N) {
            uint4 v = *(const uint4*)(ldsT + row16 * 128 + col8);
            *(uint4*)(hout + (long)m * 128 + col8) = v;
            if (agg_out)
                *(uint4*)(agg_out + (long)m * 128 + col8) = v;
        }
    }
}

// ---------------- pooling + final linear ----------------

__global__ void pool_accum(const unsigned short* __restrict__ h,
                           const int* __restrict__ batch,
                           float* __restrict__ gsums,   // [8,128] f32
                           float* __restrict__ gcnt,    // [8]
                           int N) {
    int t = blockIdx.x * blockDim.x + threadIdx.x;
    int n = t >> 5;
    if (n >= N) return;
    int lane = t & 31;
    int g = batch[n];
    int f = lane * 4;
    uint2 v = *(const uint2*)(h + (long)n * 128 + f);
    float* gs = gsums + g * 128 + f;
    unsafeAtomicAdd(gs + 0, bf2f((unsigned short)(v.x & 0xffff)));
    unsafeAtomicAdd(gs + 1, bf2f((unsigned short)(v.x >> 16)));
    unsafeAtomicAdd(gs + 2, bf2f((unsigned short)(v.y & 0xffff)));
    unsafeAtomicAdd(gs + 3, bf2f((unsigned short)(v.y >> 16)));
    if (lane == 0) unsafeAtomicAdd(&gcnt[g], 1.0f);
}

__global__ void final_lin(const float* __restrict__ gsums,
                          const float* __restrict__ gcnt,
                          const float* __restrict__ Wlin,  // [128,5]
                          const float* __restrict__ blin,  // [5]
                          float* __restrict__ out) {
    int t = threadIdx.x;
    if (t >= 8 * 5) return;
    int g = t / 5, cc = t % 5;
    float inv = 1.0f / fmaxf(gcnt[g], 1.0f);
    float acc = blin[cc];
    const float* gs = gsums + g * 128;
#pragma unroll 8
    for (int f = 0; f < 128; ++f) acc = fmaf(gs[f] * inv, Wlin[f * 5 + cc], acc);
    out[t] = acc;
}

// ---------------------------------------------------------------------------

extern "C" void kernel_launch(void* const* d_in, const int* in_sizes, int n_in,
                              void* d_out, int out_size, void* d_ws, size_t ws_size,
                              hipStream_t stream) {
    const float* x     = (const float*)d_in[0];
    const int*   ei    = (const int*)  d_in[1];
    const float* ea    = (const float*)d_in[2];
    const int*   batch = (const int*)  d_in[3];
    const float* We1 = (const float*)d_in[4],  *be1 = (const float*)d_in[5];
    const float* W1  = (const float*)d_in[6],  *b1  = (const float*)d_in[7];
    const float* We2 = (const float*)d_in[8],  *be2 = (const float*)d_in[9];
    const float* W2  = (const float*)d_in[10], *b2  = (const float*)d_in[11];
    const float* We3 = (const float*)d_in[12], *be3 = (const float*)d_in[13];
    const float* W3  = (const float*)d_in[14], *b3  = (const float*)d_in[15];
    const float* Wlin = (const float*)d_in[16], *blin = (const float*)d_in[17];

    const int N = in_sizes[0] / 7;
    const int E = in_sizes[2];
    const int* srcI = ei;
    const int* dstI = ei + E;

    // workspace layout (bf16 feature buffers)
    unsigned short* aggB = (unsigned short*)d_ws;          // N*128 bf16
    unsigned short* hA   = aggB + (size_t)N * 128;         // N*128 bf16
    unsigned short* hB   = hA   + (size_t)N * 128;         // N*128 bf16
    unsigned short* Wb   = hB   + (size_t)N * 128;         // 128*128 bf16
    float* gsums = (float*)(Wb + 128 * 128);               // 8*128 f32
    float* gcnt  = gsums + 8 * 128;                        // 8 f32
    // layer-1 f32 scatter scratch: reuse hB's region (N*7 f32 = 2.8MB < N*128*2B;
    // hB is not live until layer-2's GEMM fully overwrites it). No aliasing with
    // aggB/hA, so gemm7's dual-write into aggB is race-free.
    float* agg7s = (float*)hB;

    const int T = 256;
    auto cdiv = [](int a, int b) { return (a + b - 1) / b; };

    // ---- layer 1 (7 -> 128), f32 scatter ----
    copy_f32<<<cdiv(N * 7, T), T, 0, stream>>>(agg7s, x, N * 7);
    edge_scatter7<<<cdiv(E, T), T, 0, stream>>>(x, srcI, dstI, ea, We1, be1, agg7s, E);
    gemm7_relu<<<cdiv(N * 128, T), T, 0, stream>>>(agg7s, W1, b1, hA, aggB, N);

    // ---- layer 2 (128 -> 128), bf16 ----
    edge_scatter128_k<<<cdiv(E, 8), T, 0, stream>>>(hA, srcI, dstI, ea, We2, be2, aggB, E);
    stage_w_bf16<<<cdiv(128 * 128, T), T, 0, stream>>>(W2, Wb);
    // dual-write: hB for the edge gather, and aggB (aliasing input) as layer-3 init
    gemm128_wmma_relu<<<cdiv(N, 16), T, 0, stream>>>(aggB, Wb, b2, hB, aggB, N);

    // ---- layer 3 (128 -> 128), bf16 ----
    edge_scatter128_k<<<cdiv(E, 8), T, 0, stream>>>(hB, srcI, dstI, ea, We3, be3, aggB, E);
    stage_w_bf16<<<cdiv(128 * 128, T), T, 0, stream>>>(W3, Wb);
    gemm128_wmma_relu<<<cdiv(N, 16), T, 0, stream>>>(aggB, Wb, b3, hA, (unsigned short*)nullptr, N);

    // ---- global mean pool + final linear ----
    zero_f32<<<cdiv(8 * 128 + 8, T), T, 0, stream>>>(gsums, 8 * 128 + 8);
    pool_accum<<<cdiv(N * 32, T), T, 0, stream>>>(hA, batch, gsums, gcnt, N);
    final_lin<<<1, 64, 0, stream>>>(gsums, gcnt, Wlin, blin, (float*)d_out);
}
